// GaugeBlock_55654186221827
// MI455X (gfx1250) — compile-verified, matrix operand order checked by reading, not run on hardware
//
#include <hip/hip_runtime.h>
#include <hip/hip_bf16.h>
#include <math.h>

// ---------------------------------------------------------------------------
// Problem constants (from reference): B=4, T=2048, D=2048, H=16, HD=128
// ---------------------------------------------------------------------------
#define BB 4
#define TT 2048
#define DD 2048
#define HH 16
#define HDD 128
#define DHALF (DD / 2)

typedef __attribute__((ext_vector_type(16))) __bf16 v16bf;
typedef __attribute__((ext_vector_type(8)))  __bf16 v8bf;
typedef __attribute__((ext_vector_type(8)))  float  v8f;
typedef int v4i __attribute__((vector_size(16)));   // matches async-LDS builtin param

#define AS1 __attribute__((address_space(1)))
#define AS3 __attribute__((address_space(3)))

// CDNA5 async global->LDS copy path (ASYNCcnt tracked), guarded so the file
// still compiles on toolchains without the builtins.
#if defined(__has_builtin)
#if __has_builtin(__builtin_amdgcn_global_load_async_to_lds_b128) && \
    __has_builtin(__builtin_amdgcn_s_wait_asynccnt)
#define USE_ASYNC_LDS 1
#endif
#endif
#ifndef USE_ASYNC_LDS
#define USE_ASYNC_LDS 0
#endif

__device__ __forceinline__ v16bf combine16(v8bf lo, v8bf hi) {
    union { v16bf v; v8bf h[2]; } u;
    u.h[0] = lo; u.h[1] = hi;
    return u.v;
}

__device__ __forceinline__ v8f vzero8f() {
    v8f z = {0.f, 0.f, 0.f, 0.f, 0.f, 0.f, 0.f, 0.f};
    return z;
}

// ---------------------------------------------------------------------------
// Elementwise f32 -> bf16 conversion
// ---------------------------------------------------------------------------
__global__ __launch_bounds__(256) void cvt_f32_bf16(const float* __restrict__ src,
                                                    __bf16* __restrict__ dst,
                                                    long long n) {
    long long i = (long long)blockIdx.x * blockDim.x + threadIdx.x;
    if (i < n) dst[i] = (__bf16)src[i];
}

// ---------------------------------------------------------------------------
// Tiled bf16 WMMA GEMM:  C[M,N] (f32) = A[M,K] (bf16) * W[N,K]^T (bf16)
// Block: 256 threads = 8 waves. Block tile 128(M) x 64(N). K step 32.
// W tiles double-buffered in LDS via global_load_async_to_lds_b128 (ASYNCcnt)
// when available, so the HBM/L2 fetch of tile k+1 overlaps the WMMAs on tile
// k; falls back to the synchronous load/ds_store pipeline otherwise.
// ---------------------------------------------------------------------------
__global__ __launch_bounds__(256) void gemm_bf16_wmma(const __bf16* __restrict__ A,
                                                      const __bf16* __restrict__ W,
                                                      float* __restrict__ C,
                                                      int M, int N, int K) {
    __shared__ __bf16 wt[2][64][40];   // [buf][n][k] tile, padded stride 80B

    const int tid  = threadIdx.x;
    const int wv   = tid >> 5;
    const int lane = tid & 31;
    const int hf   = lane >> 4;     // half-wave select (K-run select)
    const int lp   = lane & 15;     // row (A) / col (B,C,D) within tile

    const int mBase = blockIdx.x * 128 + wv * 16;
    const int nBase = blockIdx.y * 64;

    v8f acc[4];
    for (int i = 0; i < 4; ++i) acc[i] = vzero8f();

    const int stage_n = tid >> 2;          // 0..63
    const int stage_k = (tid & 3) * 8;     // 0,8,16,24

    // One K-step of compute from LDS tile `buf` at global K offset k0.
    auto compute_step = [&](const __bf16 (&tile)[64][40], int k0) {
        // A fragment: row = mBase+lp, K runs [k0+8*hf, +8) and [k0+16+8*hf, +8)
        const __bf16* arow = &A[(size_t)(mBase + lp) * K + k0];
        v8bf alo = *reinterpret_cast<const v8bf*>(arow + 8 * hf);
        v8bf ahi = *reinterpret_cast<const v8bf*>(arow + 16 + 8 * hf);
        v16bf af = combine16(alo, ahi);
        #pragma unroll
        for (int nt = 0; nt < 4; ++nt) {
            v8bf blo = *reinterpret_cast<const v8bf*>(&tile[nt * 16 + lp][8 * hf]);
            v8bf bhi = *reinterpret_cast<const v8bf*>(&tile[nt * 16 + lp][16 + 8 * hf]);
            v16bf bf_ = combine16(blo, bhi);
            acc[nt] = __builtin_amdgcn_wmma_f32_16x16x32_bf16(
                false, af, false, bf_, (short)0, acc[nt], false, false);
        }
    };

#if USE_ASYNC_LDS
    // prologue: async-stage tile 0 into buffer 0
    __builtin_amdgcn_global_load_async_to_lds_b128(
        (AS1 v4i*)&W[(size_t)(nBase + stage_n) * K + stage_k],
        (AS3 v4i*)&wt[0][stage_n][stage_k], 0, 0);
    __builtin_amdgcn_s_wait_asynccnt(0);
    __syncthreads();

    int buf = 0;
    for (int k0 = 0; k0 < K; k0 += 32) {
        if (k0 + 32 < K) {
            // overlap: fetch next W tile into the other buffer while computing
            __builtin_amdgcn_global_load_async_to_lds_b128(
                (AS1 v4i*)&W[(size_t)(nBase + stage_n) * K + (k0 + 32) + stage_k],
                (AS3 v4i*)&wt[buf ^ 1][stage_n][stage_k], 0, 0);
        }
        compute_step(wt[buf], k0);
        __builtin_amdgcn_s_wait_asynccnt(0);
        __syncthreads();
        buf ^= 1;
    }
#else
    for (int k0 = 0; k0 < K; k0 += 32) {
        v8bf wv8 = *reinterpret_cast<const v8bf*>(
            &W[(size_t)(nBase + stage_n) * K + k0 + stage_k]);
        *reinterpret_cast<v8bf*>(&wt[0][stage_n][stage_k]) = wv8;
        __syncthreads();
        compute_step(wt[0], k0);
        __syncthreads();
    }
#endif

    // D-frag store: VGPR j -> row j (lanes 0-15) / j+8 (lanes 16-31), col = lp
    #pragma unroll
    for (int nt = 0; nt < 4; ++nt) {
        #pragma unroll
        for (int j = 0; j < 8; ++j) {
            int row = mBase + j + 8 * hf;
            C[(size_t)row * N + nBase + nt * 16 + lp] = acc[nt][j];
        }
    }
}

// ---------------------------------------------------------------------------
// In-place cumulative sum over T for ang (B, T, D/2). One thread per (b, c).
// ---------------------------------------------------------------------------
__global__ __launch_bounds__(256) void cumsum_time(float* __restrict__ ang) {
    int idx = blockIdx.x * blockDim.x + threadIdx.x;
    if (idx >= BB * DHALF) return;
    int b = idx / DHALF;
    int c = idx % DHALF;
    float acc = 0.f;
    for (int t = 0; t < TT; ++t) {
        size_t off = ((size_t)b * TT + t) * DHALF + c;
        acc += ang[off];
        ang[off] = acc;
    }
}

// ---------------------------------------------------------------------------
// Forward Givens rotation: x (f32) rotated pairwise by ang -> xg (bf16)
// ---------------------------------------------------------------------------
__global__ __launch_bounds__(256) void givens_fwd(const float* __restrict__ x,
                                                  const float* __restrict__ ang,
                                                  __bf16* __restrict__ xg) {
    long long idx = (long long)blockIdx.x * blockDim.x + threadIdx.x;
    if (idx >= (long long)BB * TT * DHALF) return;
    long long rowp = idx / DHALF;   // b*T + t
    int p = (int)(idx % DHALF);
    float th = ang[idx];
    float s = sinf(th), c = cosf(th);
    const float* xr = x + rowp * DD;
    float xe = xr[2 * p], xo = xr[2 * p + 1];
    __bf16* xo_ = xg + rowp * DD;
    xo_[2 * p]     = (__bf16)(c * xe - s * xo);
    xo_[2 * p + 1] = (__bf16)(s * xe + c * xo);
}

// ---------------------------------------------------------------------------
// Inverse Givens rotation: yp (f32) -> out (f32), s -> -s
// ---------------------------------------------------------------------------
__global__ __launch_bounds__(256) void givens_inv(const float* __restrict__ yp,
                                                  const float* __restrict__ ang,
                                                  float* __restrict__ out) {
    long long idx = (long long)blockIdx.x * blockDim.x + threadIdx.x;
    if (idx >= (long long)BB * TT * DHALF) return;
    long long rowp = idx / DHALF;
    int p = (int)(idx % DHALF);
    float th = ang[idx];
    float s = -sinf(th), c = cosf(th);
    const float* yr = yp + rowp * DD;
    float xe = yr[2 * p], xo = yr[2 * p + 1];
    float* orow = out + rowp * DD;
    orow[2 * p]     = c * xe - s * xo;
    orow[2 * p + 1] = s * xe + c * xo;
}

// ---------------------------------------------------------------------------
// RoPE + pack: qkv f32 (B,T,3D) -> q_pk,k_pk bf16 (B,H,T,HD), v_pk bf16
// transposed (B,H,HD,T) so attention PV B-frags are contiguous in T.
// ---------------------------------------------------------------------------
__global__ __launch_bounds__(256) void rope_pack(const float* __restrict__ qkv,
                                                 const float* __restrict__ cosb,
                                                 const float* __restrict__ sinb,
                                                 __bf16* __restrict__ qpk,
                                                 __bf16* __restrict__ kpk,
                                                 __bf16* __restrict__ vpk) {
    long long idx = (long long)blockIdx.x * blockDim.x + threadIdx.x;
    if (idx >= (long long)BB * HH * TT * (HDD / 2)) return;
    int j  = (int)(idx & 63);
    long long r = idx >> 6;
    int t  = (int)(r % TT);
    int bh = (int)(r / TT);
    int b  = bh / HH;
    int h  = bh % HH;

    float c = cosb[t * 64 + j];
    float s = sinb[t * 64 + j];

    const float* base = qkv + ((size_t)b * TT + t) * (3 * DD) + h * HDD;
    {
        float x1 = base[j], x2 = base[j + 64];
        __bf16* q = qpk + ((size_t)bh * TT + t) * HDD;
        q[j]      = (__bf16)(x1 * c + x2 * s);
        q[j + 64] = (__bf16)(-x1 * s + x2 * c);
    }
    {
        const float* kb = base + DD;
        float x1 = kb[j], x2 = kb[j + 64];
        __bf16* k = kpk + ((size_t)bh * TT + t) * HDD;
        k[j]      = (__bf16)(x1 * c + x2 * s);
        k[j + 64] = (__bf16)(-x1 * s + x2 * c);
    }
    {
        const float* vb = base + 2 * DD;
        __bf16* v = vpk + (size_t)bh * HDD * TT;
        v[(size_t)j * TT + t]        = (__bf16)vb[j];
        v[(size_t)(j + 64) * TT + t] = (__bf16)vb[j + 64];
    }
}

// ---------------------------------------------------------------------------
// Causal flash attention. blockDim=128 (4 waves); each wave owns one 16-row
// q tile and streams keys in steps of 32 with online softmax.
// ---------------------------------------------------------------------------
__global__ __launch_bounds__(128) void attn_flash(const __bf16* __restrict__ qpk,
                                                  const __bf16* __restrict__ kpk,
                                                  const __bf16* __restrict__ vpk,
                                                  __bf16* __restrict__ ybf,
                                                  float scale) {
    __shared__ float  sS[4][16][33];
    __shared__ __bf16 sP[4][16][40];
    __shared__ float  sM[4][16];
    __shared__ float  sL[4][16];
    __shared__ float  sAl[4][16];

    const int tid  = threadIdx.x;
    const int wv   = tid >> 5;
    const int lane = tid & 31;
    const int hf   = lane >> 4;
    const int lp   = lane & 15;

    const int bh = blockIdx.x / 32;           // (b*H + h)
    const int b  = bh / HH;
    const int h  = bh % HH;
    const int qt = (blockIdx.x % 32) * 4 + wv;
    const int qb = qt * 16;

    // Q fragments (16 x 128 split into 4 K-groups of 32)
    const __bf16* Q = qpk + (size_t)bh * TT * HDD;
    v16bf qf[4];
    {
        const __bf16* qrow = &Q[(size_t)(qb + lp) * HDD];
        #pragma unroll
        for (int g = 0; g < 4; ++g) {
            v8bf lo = *reinterpret_cast<const v8bf*>(qrow + g * 32 + 8 * hf);
            v8bf hi = *reinterpret_cast<const v8bf*>(qrow + g * 32 + 16 + 8 * hf);
            qf[g] = combine16(lo, hi);
        }
    }

    v8f o[8];
    #pragma unroll
    for (int f = 0; f < 8; ++f) o[f] = vzero8f();

    if (lane < 16) { sM[wv][lane] = -1e30f; sL[wv][lane] = 0.f; }
    asm volatile("s_wait_dscnt 0" ::: "memory");

    const __bf16* Kb = kpk + (size_t)bh * TT * HDD;
    const __bf16* Vb = vpk + (size_t)bh * HDD * TT;

    for (int kb = 0; kb < qb + 16; kb += 32) {
        // ---- scores: two 16x16 tiles covering keys [kb, kb+32)
        #pragma unroll
        for (int st = 0; st < 2; ++st) {
            v8f s = vzero8f();
            const __bf16* krow = &Kb[(size_t)(kb + st * 16 + lp) * HDD];
            #pragma unroll
            for (int g = 0; g < 4; ++g) {
                v8bf lo = *reinterpret_cast<const v8bf*>(krow + g * 32 + 8 * hf);
                v8bf hi = *reinterpret_cast<const v8bf*>(krow + g * 32 + 16 + 8 * hf);
                v16bf bf_ = combine16(lo, hi);
                s = __builtin_amdgcn_wmma_f32_16x16x32_bf16(
                    false, qf[g], false, bf_, (short)0, s, false, false);
            }
            int kidx = kb + st * 16 + lp;
            #pragma unroll
            for (int j = 0; j < 8; ++j) {
                int r = j + 8 * hf;
                float val = s[j] * scale;
                if (kidx > qb + r) val = -3.0e38f;
                sS[wv][r][st * 16 + lp] = val;
            }
        }
        asm volatile("s_wait_dscnt 0" ::: "memory");

        // ---- online softmax: lane r < 16 owns row r
        if (lane < 16) {
            const int r = lane;
            float mOld = sM[wv][r];
            float mNew = mOld;
            #pragma unroll 4
            for (int c = 0; c < 32; ++c) mNew = fmaxf(mNew, sS[wv][r][c]);
            float alpha = __expf(mOld - mNew);
            float sum = 0.f;
            #pragma unroll 4
            for (int c = 0; c < 32; ++c) {
                float p = __expf(sS[wv][r][c] - mNew);
                sum += p;
                sP[wv][r][c] = (__bf16)p;
            }
            sL[wv][r] = sL[wv][r] * alpha + sum;
            sM[wv][r] = mNew;
            sAl[wv][r] = alpha;
        }
        asm volatile("s_wait_dscnt 0" ::: "memory");

        // ---- rescale accumulator (8 LDS reads, reused across 8 O-frags)
        float al[8];
        #pragma unroll
        for (int j = 0; j < 8; ++j) al[j] = sAl[wv][j + 8 * hf];
        #pragma unroll
        for (int f = 0; f < 8; ++f) {
            #pragma unroll
            for (int j = 0; j < 8; ++j) o[f][j] *= al[j];
        }

        // ---- P fragment (16 x 32) from LDS in A-frag layout
        v8bf plo = *reinterpret_cast<const v8bf*>(&sP[wv][lp][8 * hf]);
        v8bf phi = *reinterpret_cast<const v8bf*>(&sP[wv][lp][16 + 8 * hf]);
        v16bf pf = combine16(plo, phi);

        // ---- O += P * V   (V^T layout makes B-frag loads contiguous in T)
        #pragma unroll
        for (int f = 0; f < 8; ++f) {
            const __bf16* vrow = &Vb[(size_t)(f * 16 + lp) * TT + kb];
            v8bf vlo = *reinterpret_cast<const v8bf*>(vrow + 8 * hf);
            v8bf vhi = *reinterpret_cast<const v8bf*>(vrow + 16 + 8 * hf);
            v16bf vf = combine16(vlo, vhi);
            o[f] = __builtin_amdgcn_wmma_f32_16x16x32_bf16(
                false, pf, false, vf, (short)0, o[f], false, false);
        }
    }

    // ---- normalize and store y (B,T,D) as bf16 for the projection GEMM
    float ln[8];
    #pragma unroll
    for (int j = 0; j < 8; ++j) ln[j] = sL[wv][j + 8 * hf];
    #pragma unroll
    for (int f = 0; f < 8; ++f) {
        #pragma unroll
        for (int j = 0; j < 8; ++j) {
            int r = j + 8 * hf;
            float val = o[f][j] / ln[j];
            ybf[((size_t)b * TT + qb + r) * DD + h * HDD + f * 16 + lp] = (__bf16)val;
        }
    }
}

// ---------------------------------------------------------------------------
// Host-side launcher
// ---------------------------------------------------------------------------
extern "C" void kernel_launch(void* const* d_in, const int* in_sizes, int n_in,
                              void* d_out, int out_size, void* d_ws, size_t ws_size,
                              hipStream_t stream) {
    (void)in_sizes; (void)n_in; (void)out_size; (void)ws_size;

    const float* x    = (const float*)d_in[0];
    const float* cosb = (const float*)d_in[1];
    const float* sinb = (const float*)d_in[2];
    const float* wA   = (const float*)d_in[3];
    const float* wQKV = (const float*)d_in[4];
    const float* wP   = (const float*)d_in[5];
    float* out = (float*)d_out;

    char* ws = (char*)d_ws;
    const size_t MB = 1ull << 20;
    float*  ang    = (float*) (ws + 0 * MB);      // 32 MB  (B,T,D/2)
    __bf16* wA_b   = (__bf16*)(ws + 32 * MB);     //  4 MB
    __bf16* wQKV_b = (__bf16*)(ws + 36 * MB);     // 24 MB
    __bf16* wP_b   = (__bf16*)(ws + 60 * MB);     //  8 MB
    __bf16* x_b    = (__bf16*)(ws + 68 * MB);     // 32 MB  (reused later as ybf)
    __bf16* xg     = (__bf16*)(ws + 100 * MB);    // 32 MB
    float*  qkv    = (float*) (ws + 132 * MB);    // 192 MB (reused later as yproj)
    __bf16* qpk    = (__bf16*)(ws + 324 * MB);    // 32 MB
    __bf16* kpk    = (__bf16*)(ws + 356 * MB);    // 32 MB
    __bf16* vpk    = (__bf16*)(ws + 388 * MB);    // 32 MB
    __bf16* ybf    = x_b;                         // reuse (x_b dead after gemm1)
    float*  yproj  = qkv;                         // reuse (qkv dead after rope_pack)

    const long long nX = (long long)BB * TT * DD;

    // 1) bf16 conversions for WMMA operands
    cvt_f32_bf16<<<(unsigned)((nX + 255) / 256), 256, 0, stream>>>(x, x_b, nX);
    cvt_f32_bf16<<<(unsigned)((DHALF * (long long)DD + 255) / 256), 256, 0, stream>>>(
        wA, wA_b, (long long)DHALF * DD);
    cvt_f32_bf16<<<(unsigned)((3ll * DD * DD + 255) / 256), 256, 0, stream>>>(
        wQKV, wQKV_b, 3ll * DD * DD);
    cvt_f32_bf16<<<(unsigned)(((long long)DD * DD + 255) / 256), 256, 0, stream>>>(
        wP, wP_b, (long long)DD * DD);

    const int M = BB * TT;  // 8192

    // 2) angles_raw = x @ w_angles^T   (M x 1024, K=2048)
    gemm_bf16_wmma<<<dim3(M / 128, DHALF / 64), 256, 0, stream>>>(
        x_b, wA_b, ang, M, DHALF, DD);

    // 3) cumsum over time
    cumsum_time<<<(BB * DHALF + 255) / 256, 256, 0, stream>>>(ang);

    // 4) forward Givens (writes bf16 activations for next GEMM)
    givens_fwd<<<(unsigned)(((long long)BB * TT * DHALF + 255) / 256), 256, 0, stream>>>(
        x, ang, xg);

    // 5) qkv = x_g @ w_attn^T  (M x 6144, K=2048)
    gemm_bf16_wmma<<<dim3(M / 128, (3 * DD) / 64), 256, 0, stream>>>(
        xg, wQKV_b, qkv, M, 3 * DD, DD);

    // 6) RoPE + pack q/k (B,H,T,HD) and v transposed (B,H,HD,T)
    rope_pack<<<(unsigned)(((long long)BB * HH * TT * (HDD / 2) + 255) / 256), 256, 0, stream>>>(
        qkv, cosb, sinb, qpk, kpk, vpk);

    // 7) causal flash attention -> ybf (B,T,D) bf16
    const float scale = 1.0f / sqrtf((float)HDD);
    attn_flash<<<BB * HH * (TT / 16) / 4, 128, 0, stream>>>(qpk, kpk, vpk, ybf, scale);

    // 8) y @ w_proj^T  (M x 2048, K=2048)
    gemm_bf16_wmma<<<dim3(M / 128, DD / 64), 256, 0, stream>>>(
        ybf, wP_b, yproj, M, DD, DD);

    // 9) inverse Givens -> d_out (f32)
    givens_inv<<<(unsigned)(((long long)BB * TT * DHALF + 255) / 256), 256, 0, stream>>>(
        yproj, ang, out);
}
